// deform_conv3d_naive_74483322847775
// MI455X (gfx1250) — matrix operand order repaired and checked by reading, not compiled, not run on hardware
//
#include <hip/hip_runtime.h>
#include <math.h>

typedef _Float16 half_t;
typedef __attribute__((ext_vector_type(16))) _Float16 v16h;
typedef __attribute__((ext_vector_type(8)))  _Float16 v8h;
typedef __attribute__((ext_vector_type(8)))  float    v8f;

namespace {
constexpr int NB   = 2;
constexpr int CIN  = 64;
constexpr int COUT = 64;
constexpr int DD = 8, HH = 56, WW = 56;
constexpr int KT  = 27;                  // real taps (3x3x3)
constexpr int KP  = 32;                  // padded taps per channel (WMMA K)
constexpr int DHW = DD * HH * WW;        // 25088 (divisible by 64)
constexpr int TILE_P = 64;               // output positions per block
constexpr int LDK = 40;                  // padded B-tile row stride in halves (20 dwords)
constexpr int CTE = KP * TILE_P;         // 2048 corner-table entries (incl. zero-weight pad taps)

struct Corner { int idx; float w; };     // 8 bytes

constexpr size_t SMEM_BYTES = sizeof(Corner) * CTE * 8          // 131072 B corner table
                            + 2 * sizeof(half_t) * TILE_P * LDK; //  10240 B double-buffered B tile
constexpr size_t WF_HALVES = (size_t)CIN * COUT * KP;            // 131072 f16 = 256 KB in d_ws
} // namespace

// One-shot: weight (Cout,Cin,27) fp32 -> padded f16 A matrix wf[c][co][32] (zero pad taps)
__global__ __launch_bounds__(256)
void convert_weights(const float* __restrict__ wgt, half_t* __restrict__ wf) {
    const int e  = blockIdx.x * 256 + threadIdx.x;   // e < CIN*COUT*32
    const int t  = e & 31;
    const int co = (e >> 5) & 63;
    const int c  = e >> 11;
    wf[e] = (t < KT) ? (half_t)wgt[(co * CIN + c) * KT + t] : (half_t)0.0f;
}

__global__ __launch_bounds__(256)
void deform_conv3d_wmma(const float* __restrict__ x,
                        const float* __restrict__ off,
                        const half_t* __restrict__ wf,
                        float* __restrict__ out) {
    extern __shared__ unsigned char smem[];
    Corner* ctab = reinterpret_cast<Corner*>(smem);                 // [8 corners][CTE]
    half_t* bt0  = reinterpret_cast<half_t*>(smem + sizeof(Corner) * CTE * 8);
    half_t* bt1  = bt0 + TILE_P * LDK;

    const int tid      = threadIdx.x;
    const int pos_base = blockIdx.x * TILE_P;
    const int n        = pos_base / DHW;
    const int rem_base = pos_base - n * DHW;
    const float* xb  = x   + (size_t)n * CIN * DHW;
    const float* ofb = off + (size_t)n * (KT * 3) * DHW;

    // ---- Phase 1: corner table, branch-free, exactly 8 unrolled passes of 256 threads.
    // Pad taps (t >= 27) get weight 0 (idx of tap 0) so the hot loop needs no conditionals.
#pragma unroll
    for (int i = 0; i < 8; ++i) {
        const int e = tid + i * 256;          // e = t*64 + p
        const int p = e & (TILE_P - 1);
        const int t = e >> 6;
        const int rem = rem_base + p;
        const int d   = rem / (HH * WW);
        const int hw  = rem - d * (HH * WW);
        const int h   = hw / WW;
        const int w   = hw - h * WW;
        const int tc  = (t < KT) ? t : 0;     // clamp instead of branch
        const float tv = (t < KT) ? 1.0f : 0.0f;
        const int kz = tc / 9, kr = tc - kz * 9;
        const int ky = kr / 3, kx = kr - ky * 3;
        const float z  = (float)(d - 1 + kz) + ofb[(tc * 3 + 0) * DHW + rem];
        const float y  = (float)(h - 1 + ky) + ofb[(tc * 3 + 1) * DHW + rem];
        const float xx = (float)(w - 1 + kx) + ofb[(tc * 3 + 2) * DHW + rem];
        const float zf = floorf(z), yf = floorf(y), xf = floorf(xx);
        const int z0 = (int)zf, y0 = (int)yf, x0 = (int)xf;
        const float fz = z - zf, fy = y - yf, fx = xx - xf;
#pragma unroll
        for (int j = 0; j < 8; ++j) {
            const int dz = j >> 2, dy = (j >> 1) & 1, dx = j & 1;
            const int zi = z0 + dz, yi = y0 + dy, xi = x0 + dx;
            const bool v = (zi >= 0) & (zi < DD) & (yi >= 0) & (yi < HH) & (xi >= 0) & (xi < WW);
            const int zc = zi < 0 ? 0 : (zi > DD - 1 ? DD - 1 : zi);
            const int yc = yi < 0 ? 0 : (yi > HH - 1 ? HH - 1 : yi);
            const int xc = xi < 0 ? 0 : (xi > WW - 1 ? WW - 1 : xi);
            const float wz = dz ? fz : 1.0f - fz;
            const float wy = dy ? fy : 1.0f - fy;
            const float wx = dx ? fx : 1.0f - fx;
            Corner cc;
            cc.idx = v ? ((zc * HH + yc) * WW + xc) : 0;
            cc.w   = (v ? wz * wy * wx : 0.0f) * tv;
            ctab[j * CTE + e] = cc;           // corner-major: 8B lane stride, conflict-free
        }
    }
    __syncthreads();

    // ---- GEMM assignment: 8 waves, each 16 couts x 32 positions (two 16x16 subtiles)
    const int lane  = tid & 31;
    const int wv    = tid >> 5;
    const int mi    = wv & 3;
    const int npair = wv >> 2;
    const int l15   = lane & 15;
    const int khi   = lane >> 4;

    const int m     = mi * 16 + l15;
    const int p0    = npair * 32 + l15;
    const int boff0 = p0 * LDK + khi * 16;
    const int boff1 = (p0 + 16) * LDK + khi * 16;
    const int aoffL = m * KP + khi * 16;       // global A fragment offsets (halves)
    const int aoffH = m * KP + khi * 16 + 8;

    v8f acc0 = {0.f, 0.f, 0.f, 0.f, 0.f, 0.f, 0.f, 0.f};
    v8f acc1 = acc0;

    // ---- Phase 2: K loop, one chunk (32 padded taps) per input channel.
    for (int c = 0; c < CIN; ++c) {
        half_t* btw = (c & 1) ? bt1 : bt0;
        const float* xc = xb + c * DHW;

        // B tile: branch-free deformable im2col, 8 unconditional gathers per element
#pragma unroll
        for (int i = 0; i < 8; ++i) {
            const int e = tid + i * 256;      // e = t*64 + p
            const int p = e & (TILE_P - 1);
            const int t = e >> 6;
            float a = 0.0f;
#pragma unroll
            for (int j = 0; j < 8; ++j) {
                const Corner cc = ctab[j * CTE + e];
                a += xc[cc.idx] * cc.w;
            }
            btw[p * LDK + t] = (half_t)a;
        }

        // A fragment straight from L2-resident pre-converted f16 weights (no LDS staging)
        const half_t* wfc = wf + (size_t)c * (COUT * KP);
        const v8h alo = *(const v8h*)(wfc + aoffL);
        const v8h ahi = *(const v8h*)(wfc + aoffH);
        const v16h A = __builtin_shufflevector(alo, ahi, 0,1,2,3,4,5,6,7,8,9,10,11,12,13,14,15);

        __syncthreads();   // single barrier per chunk (double-buffered B tile)

        const v8h b0l = *(const v8h*)(btw + boff0);
        const v8h b0h = *(const v8h*)(btw + boff0 + 8);
        const v16h B0 = __builtin_shufflevector(b0l, b0h, 0,1,2,3,4,5,6,7,8,9,10,11,12,13,14,15);
        const v8h b1l = *(const v8h*)(btw + boff1);
        const v8h b1h = *(const v8h*)(btw + boff1 + 8);
        const v16h B1 = __builtin_shufflevector(b1l, b1h, 0,1,2,3,4,5,6,7,8,9,10,11,12,13,14,15);

        acc0 = __builtin_amdgcn_wmma_f32_16x16x32_f16(false, A, false, B0, (short)0, acc0, false, false);
        acc1 = __builtin_amdgcn_wmma_f32_16x16x32_f16(false, A, false, B1, (short)0, acc1, false, false);
    }

    // ---- store: VGPR r holds (M = r + 8*khi, N = l15); coalesced along positions
    const size_t obase = (size_t)n * COUT * DHW + rem_base;
    const int coutb = mi * 16 + khi * 8;
    const int col0  = npair * 32 + l15;
#pragma unroll
    for (int r = 0; r < 8; ++r) {
        out[obase + (size_t)(coutb + r) * DHW + col0]      = acc0[r];
        out[obase + (size_t)(coutb + r) * DHW + col0 + 16] = acc1[r];
    }
}

extern "C" void kernel_launch(void* const* d_in, const int* in_sizes, int n_in,
                              void* d_out, int out_size, void* d_ws, size_t ws_size,
                              hipStream_t stream) {
    (void)in_sizes; (void)n_in; (void)out_size; (void)ws_size;
    const float* x   = (const float*)d_in[0];
    const float* off = (const float*)d_in[1];
    const float* w   = (const float*)d_in[2];
    float* out  = (float*)d_out;
    half_t* wf  = (half_t*)d_ws;   // 256 KB padded f16 weights

    convert_weights<<<dim3((int)(WF_HALVES / 256)), dim3(256), 0, stream>>>(w, wf);

    const int nblocks = (NB * DHW) / TILE_P;   // 784
    deform_conv3d_wmma<<<dim3(nblocks), dim3(256), SMEM_BYTES, stream>>>(x, off, wf, out);
}
